// CNN_58480274702719
// MI455X (gfx1250) — matrix operand order repaired and checked by reading, not compiled, not run on hardware
//
#include <hip/hip_runtime.h>

typedef __attribute__((ext_vector_type(16))) __bf16 v16bf;
typedef __attribute__((ext_vector_type(8)))  float  v8f;

__device__ __forceinline__ unsigned short f2bf_rne(float f) {
  unsigned u = __builtin_bit_cast(unsigned, f);
  unsigned r = u + 0x7FFFu + ((u >> 16) & 1u);
  return (unsigned short)(r >> 16);
}
__device__ __forceinline__ float bf2f(unsigned short h) {
  unsigned u = ((unsigned)h) << 16;
  return __builtin_bit_cast(float, u);
}

// -------------------------------------------------------------------------
// Implicit-GEMM 3x3 SAME conv, stride 1, square images (W = H = 2^lw),
// NHWC bf16 activations, weights repacked to [CoutPad][9*Cin] bf16.
//   A (16x32): 16 output pixels x 32 input channels of one tap
//   B (32x16): 16 cout columns, contiguous K per column
// Wave tile: 64(M) x 64(N) -> 16 f32 accumulators; per K-chunk:
//   4 A-fragment loads + 4 B-fragment loads (16 x b128) feed 16 WMMAs
//   (1.0 loads/WMMA vs 1.5 in the 64x32 variant -> less L0 pressure).
// Block: 8 M-waves -> 512(M) x 64(N); all waves share one B stream.
// Out-of-image taps read from a 2KB zero buffer (branchless halo).
// mode 0: bf16 NHWC (+ReLU).  mode 1: f32 NCHW logits.
// -------------------------------------------------------------------------
__global__ __launch_bounds__(256)
void conv3x3_wmma_bf16(const unsigned short* __restrict__ in,   // [W*W][Cin]
                       const unsigned short* __restrict__ w2,   // [CoutPad][9*Cin]
                       const float* __restrict__ bias,          // [Cout]
                       unsigned short* __restrict__ out_bf,
                       float* __restrict__ out_f32,
                       const unsigned short* __restrict__ zbuf, // >=2KB zeros
                       int lw, int Cin, int Cout, int relu, int mode)
{
  const int W    = 1 << lw;
  const int M    = W << lw;
  const int Ktot = 9 * Cin;
  const int lane = threadIdx.x & 31;
  const int wave = threadIdx.x >> 5;
  const int m_base = blockIdx.x * 512 + wave * 64;
  const int n_base = blockIdx.y * 64;
  const int mrow  = lane & 15;
  const int khalf = lane >> 4;

  // Per-M-subtile pixel geometry (no guards: M % 512 == 0 always)
  int pb[4], yy[4], xm[4];
#pragma unroll
  for (int i = 0; i < 4; ++i) {
    const int p = m_base + i * 16 + mrow;
    yy[i] = p >> lw;
    xm[i] = p & (W - 1);
    pb[i] = p * Cin;
  }
  // B column base pointers (always in-bounds: weights padded to CoutPad)
  const unsigned short* bcol[4];
#pragma unroll
  for (int j = 0; j < 4; ++j)
    bcol[j] = w2 + (size_t)(n_base + j * 16 + mrow) * Ktot + 16 * khalf;

  v8f acc[4][4];
#pragma unroll
  for (int i = 0; i < 4; ++i)
#pragma unroll
    for (int j = 0; j < 4; ++j)
      acc[i][j] = (v8f){0.f,0.f,0.f,0.f,0.f,0.f,0.f,0.f};

  for (int tap = 0; tap < 9; ++tap) {
    const int dy   = tap / 3 - 1;
    const int dx   = tap - (tap / 3) * 3 - 1;
    const int doff = (dy * W + dx) * Cin;     // scalar per-tap address delta
    const int kb   = tap * Cin;
    // Branchless halo: select source pointer once per tap per subtile.
    const unsigned short* abase[4];
#pragma unroll
    for (int i = 0; i < 4; ++i) {
      const bool ok = ((unsigned)(yy[i] + dy) < (unsigned)W) &
                      ((unsigned)(xm[i] + dx) < (unsigned)W);
      abase[i] = ok ? (in + pb[i] + doff) : zbuf;
    }
    for (int c0 = 0; c0 < Cin; c0 += 32) {
      union { v16bf v; uint4 q[2]; } B[4];
#pragma unroll
      for (int j = 0; j < 4; ++j) {
        const uint4* bp = (const uint4*)(bcol[j] + kb + c0);
        B[j].q[0] = bp[0];
        B[j].q[1] = bp[1];
      }
#pragma unroll
      for (int i = 0; i < 4; ++i) {
        union { v16bf v; uint4 q[2]; } A;
        const uint4* ap = (const uint4*)(abase[i] + c0);
        A.q[0] = ap[2 * khalf];    // K = 16*khalf .. +8
        A.q[1] = ap[2 * khalf + 1]; // K = 16*khalf+8 .. +8
#pragma unroll
        for (int j = 0; j < 4; ++j)
          acc[i][j] = __builtin_amdgcn_wmma_f32_16x16x32_bf16(
              false, A.v, false, B[j].v, (short)0, acc[i][j], false, false);
      }
    }
  }

  const int mstore = 8 * khalf;
#pragma unroll
  for (int j = 0; j < 4; ++j) {
    const int ncol = n_base + j * 16 + mrow;
    if (ncol >= Cout) continue;
    const float bv = bias[ncol];
#pragma unroll
    for (int i = 0; i < 4; ++i) {
#pragma unroll
      for (int r = 0; r < 8; ++r) {
        const int m = m_base + i * 16 + r + mstore;
        float v = acc[i][j][r] + bv;
        if (relu) v = fmaxf(v, 0.0f);
        if (mode == 0) out_bf[(size_t)m * Cout + ncol] = f2bf_rne(v);
        else           out_f32[(size_t)ncol * M + m] = v;
      }
    }
  }
}

// conv1_1: Cin=3 (K=27; 0.4% of FLOPs). Fuses BGR mean subtraction and
// fp32 NCHW -> bf16 NHWC packing. One thread per (pixel, cout).
__global__ __launch_bounds__(256)
void conv1_first(const float* __restrict__ x,     // [3][512][512]
                 const float* __restrict__ wgt,   // [64][3][3][3]
                 const float* __restrict__ bias,  // [64]
                 unsigned short* __restrict__ out)// [512*512][64] bf16 NHWC
{
  const int idx = blockIdx.x * 256 + threadIdx.x;
  const int co = idx & 63;
  const int p  = idx >> 6;
  if (p >= 512 * 512) return;
  const int y = p >> 9, xx = p & 511;
  const float mean[3] = {104.008f, 116.669f, 122.675f};
  float acc = bias[co];
#pragma unroll
  for (int c = 0; c < 3; ++c) {
#pragma unroll
    for (int kh = 0; kh < 3; ++kh) {
      const int iy = y + kh - 1;
      if ((unsigned)iy >= 512u) continue;
#pragma unroll
      for (int kw = 0; kw < 3; ++kw) {
        const int ix = xx + kw - 1;
        if ((unsigned)ix >= 512u) continue;
        const float v = x[c * 512 * 512 + iy * 512 + ix] - mean[c];
        acc = fmaf(v, wgt[((co * 3 + c) * 3 + kh) * 3 + kw], acc);
      }
    }
  }
  out[(size_t)p * 64 + co] = f2bf_rne(fmaxf(acc, 0.0f));
}

// OIHW fp32 -> [CoutPad][9*Cin] bf16, k=(kh*3+kw)*Cin+cin, zero pad rows
__global__ __launch_bounds__(256)
void repack_w(const float* __restrict__ w, unsigned short* __restrict__ w2,
              int Cin, int Cout, int CoutPad)
{
  const int idx = blockIdx.x * 256 + threadIdx.x;
  const int K = 9 * Cin;
  if (idx >= CoutPad * K) return;
  const int co  = idx / K;
  const int k   = idx - co * K;
  const int tap = k / Cin;
  const int ci  = k - tap * Cin;
  const float v = (co < Cout) ? w[((size_t)co * Cin + ci) * 9 + tap] : 0.0f;
  w2[idx] = f2bf_rne(v);
}

__global__ __launch_bounds__(256)
void zero_fill(unsigned int* __restrict__ p, int n)
{
  const int i = blockIdx.x * 256 + threadIdx.x;
  if (i < n) p[i] = 0u;
}

// 2x2 max-pool, NHWC bf16
__global__ __launch_bounds__(256)
void maxpool2_nhwc(const unsigned short* __restrict__ in,
                   unsigned short* __restrict__ out, int Ho, int Wo, int C)
{
  const int idx = blockIdx.x * 256 + threadIdx.x;
  if (idx >= Ho * Wo * C) return;
  const int c = idx % C;
  const int t = idx / C;
  const int xo = t % Wo, yo = t / Wo;
  const int Wi = Wo * 2;
  const size_t b = ((size_t)(yo * 2) * Wi + xo * 2) * C + c;
  const size_t rs = (size_t)Wi * C;
  const float m = fmaxf(fmaxf(bf2f(in[b]), bf2f(in[b + C])),
                        fmaxf(bf2f(in[b + rs]), bf2f(in[b + rs + C])));
  out[idx] = f2bf_rne(m);
}

// layer channel table: 0..12 backbone, 13..17 headA, 18..22 headB,
// 23..27 headC, 28..30 headD  (weight pairs in key-sorted order)
static const int g_cin[31]  = {3,64,64,128,128,256,256,256,512,512,512,512,512,
                               256,256,128,64,32, 512,256,128,64,32,
                               512,256,128,64,32, 128,64,32};
static const int g_cout[31] = {64,64,128,128,256,256,256,512,512,512,512,512,512,
                               256,128,64,32,4, 256,128,64,32,4,
                               256,128,64,32,4, 64,32,4};

extern "C" void kernel_launch(void* const* d_in, const int* in_sizes, int n_in,
                              void* d_out, int out_size, void* d_ws, size_t ws_size,
                              hipStream_t stream) {
  (void)out_size; (void)ws_size;
  // Detect flattening order: insertion (x first) vs tree_leaves alphabetical (x last)
  int base;
  const float* x;
  if (in_sizes[0] == 3 * 512 * 512) { base = 1; x = (const float*)d_in[0]; }
  else                              { base = 0; x = (const float*)d_in[n_in - 1]; }
  auto Wg = [&](int i) { return (const float*)d_in[base + 2 * i]; };
  auto Bg = [&](int i) { return (const float*)d_in[base + 2 * i + 1]; };

  char* ws = (char*)d_ws;
  size_t off = 0;
  auto alloc = [&](size_t bytes) -> void* {
    off = (off + 255) & ~(size_t)255;
    void* p = ws + off;
    off += bytes;
    return p;
  };

  // 2KB zero halo buffer
  unsigned short* zbuf = (unsigned short*)alloc(2048);
  zero_fill<<<2, 256, 0, stream>>>((unsigned int*)zbuf, 512);

  // Repack weights to bf16 [CoutPad][9*Cin] (pad N to wave tile = 64)
  unsigned short* wp[31];
  wp[0] = nullptr;
  for (int i = 1; i < 31; ++i) {
    const int cp = (g_cout[i] + 63) & ~63;
    const size_t n = (size_t)cp * 9 * g_cin[i];
    wp[i] = (unsigned short*)alloc(n * 2);
    repack_w<<<(unsigned)((n + 255) / 256), 256, 0, stream>>>(
        Wg(i), wp[i], g_cin[i], g_cout[i], cp);
  }

  unsigned short* actA = (unsigned short*)alloc((size_t)512 * 512 * 64 * 2);
  unsigned short* actB = (unsigned short*)alloc((size_t)512 * 512 * 64 * 2);
  unsigned short* b2   = (unsigned short*)alloc((size_t)256 * 256 * 128 * 2);
  unsigned short* b3   = (unsigned short*)alloc((size_t)128 * 128 * 256 * 2);
  unsigned short* b4   = (unsigned short*)alloc((size_t)64 * 64 * 512 * 2);
  unsigned short* b5   = (unsigned short*)alloc((size_t)32 * 32 * 512 * 2);

  auto conv = [&](const unsigned short* in, int L, int Wd, int relu,
                  unsigned short* outb, float* outf) {
    const int lw = __builtin_ctz(Wd);
    dim3 grid((unsigned)((Wd * Wd) / 512), (unsigned)((g_cout[L] + 63) / 64));
    conv3x3_wmma_bf16<<<grid, 256, 0, stream>>>(in, wp[L], Bg(L), outb, outf,
                                                zbuf, lw, g_cin[L], g_cout[L],
                                                relu, outf ? 1 : 0);
  };
  auto pool = [&](const unsigned short* in, unsigned short* out, int Wo, int C) {
    const int total = Wo * Wo * C;
    maxpool2_nhwc<<<(unsigned)((total + 255) / 256), 256, 0, stream>>>(in, out, Wo, Wo, C);
  };

  // -------- backbone --------
  conv1_first<<<512 * 512 * 64 / 256, 256, 0, stream>>>(x, Wg(0), Bg(0), actA);
  conv(actA, 1, 512, 1, actB, nullptr);    // conv1_2
  pool(actB, actA, 256, 64);
  conv(actA, 2, 256, 1, actB, nullptr);    // conv2_1
  conv(actB, 3, 256, 1, b2, nullptr);      // conv2_2 -> b2
  pool(b2, actA, 128, 128);
  conv(actA, 4, 128, 1, actB, nullptr);    // conv3_1
  conv(actB, 5, 128, 1, actA, nullptr);    // conv3_2
  conv(actA, 6, 128, 1, b3, nullptr);      // conv3_3 -> b3
  pool(b3, actA, 64, 256);
  conv(actA, 7, 64, 1, actB, nullptr);     // conv4_1
  conv(actB, 8, 64, 1, actA, nullptr);     // conv4_2
  conv(actA, 9, 64, 1, b4, nullptr);       // conv4_3 -> b4
  pool(b4, actA, 32, 512);
  conv(actA, 10, 32, 1, actB, nullptr);    // conv5_1
  conv(actB, 11, 32, 1, actA, nullptr);    // conv5_2
  conv(actA, 12, 32, 1, b5, nullptr);      // conv5_3 -> b5

  float* out = (float*)d_out;
  // out0 = headC(b5) @32 -> [0, 4096)
  conv(b5, 23, 32, 1, actA, nullptr);
  conv(actA, 24, 32, 1, actB, nullptr);
  conv(actB, 25, 32, 1, actA, nullptr);
  conv(actA, 26, 32, 1, actB, nullptr);
  conv(actB, 27, 32, 0, nullptr, out + 0);
  // out1 = headB(b4) @64 -> [4096, 20480)
  conv(b4, 18, 64, 1, actA, nullptr);
  conv(actA, 19, 64, 1, actB, nullptr);
  conv(actB, 20, 64, 1, actA, nullptr);
  conv(actA, 21, 64, 1, actB, nullptr);
  conv(actB, 22, 64, 0, nullptr, out + 4096);
  // out2 = headA(b3) @128 -> [20480, 86016)
  conv(b3, 13, 128, 1, actA, nullptr);
  conv(actA, 14, 128, 1, actB, nullptr);
  conv(actB, 15, 128, 1, actA, nullptr);
  conv(actA, 16, 128, 1, actB, nullptr);
  conv(actB, 17, 128, 0, nullptr, out + 20480);
  // out3 = headD(b2) @256 -> [86016, 348160)
  conv(b2, 28, 256, 1, actA, nullptr);
  conv(actA, 29, 256, 1, actB, nullptr);
  conv(actB, 30, 256, 0, nullptr, out + 86016);
}
